// NoiseConsistencyBranch_11613591568828
// MI455X (gfx1250) — matrix operand order repaired and checked by reading, not compiled, not run on hardware
//
#include <hip/hip_runtime.h>
#include <hip/hip_bf16.h>
#include <math.h>

// ---------------------------------------------------------------------------
// NoiseConsistencyBranch for MI455X (gfx1250, wave32)
//   Pass 1: per-channel noise std        (192 blocks, LDS strip tiling)
//   Pass 2: per-image gray noise stats   (64 blocks, + LDS histogram)
//   Pass 3: features + 2-layer MLP; layer2 (64x64x32) done with
//           v_wmma_f32_16x16x32_f16 (16 waves, one WMMA tile each)
// ---------------------------------------------------------------------------

typedef __attribute__((ext_vector_type(16))) _Float16 v16h;
typedef __attribute__((ext_vector_type(8)))  float    v8f;

#define HH 512
#define WWID 512
#define HW (512 * 512)
#define STRIP 16
#define LROWS (STRIP + 4)

// ws layout (floats):
//   [0..191]    chSum      [192..383] chSumSq
//   [384..447]  graySum    [448..511] graySumSq
//   [512..3711] hist[64][50] (counts stored as float)
#define WS_CHSUM   0
#define WS_CHSQ    192
#define WS_GSUM    384
#define WS_GSQ     448
#define WS_HIST    512

__device__ __forceinline__ int refl101(int i) {
    // BORDER_REFLECT_101 for size 512, |i| <= 2 overshoot
    if (i < 0)   i = -i;
    if (i > 511) i = 1022 - i;
    return i;
}

// Gaussian 5-tap, sigma=1.0, normalized (float32 of the double values)
#define G0 0.05448868454964433f
#define G1 0.24420134200323332f
#define G2 0.4026199468942474f

template <bool GRAY>
__global__ __launch_bounds__(256)
void noise_stats_kernel(const float* __restrict__ x, float* __restrict__ ws) {
    __shared__ float    s_tile[LROWS][WWID];   // 40 KB
    __shared__ unsigned s_hist[50];
    __shared__ float    s_psum[8], s_psq[8];

    const int tid = threadIdx.x;
    const float gk[5] = {G0, G1, G2, G1, G0};

    // GRAY: one block per image (base -> 3 contiguous channels)
    // !GRAY: one block per (image,channel) plane
    const float* base = GRAY ? (x + (size_t)blockIdx.x * 3 * HW)
                             : (x + (size_t)blockIdx.x * HW);

    if (GRAY) {
        for (int i = tid; i < 50; i += 256) s_hist[i] = 0u;
    }

    float lsum = 0.f, lsq = 0.f;

    for (int y0 = 0; y0 < HH; y0 += STRIP) {
        __syncthreads();  // protect s_tile reuse across strips
        // ---- stage LROWS rows (reflected) into LDS ----
        for (int idx = tid; idx < LROWS * WWID; idx += 256) {
            const int lr  = idx >> 9;
            const int col = idx & 511;
            const int r   = refl101(y0 + lr - 2);
            float v;
            if (GRAY) {
                // uint8 truncation then cv2 RGB2GRAY with round-half-even
                const float fr = floorf(fminf(fmaxf(base[(size_t)r * WWID + col]          * 255.f, 0.f), 255.f));
                const float fg = floorf(fminf(fmaxf(base[HW + (size_t)r * WWID + col]     * 255.f, 0.f), 255.f));
                const float fb = floorf(fminf(fmaxf(base[2 * HW + (size_t)r * WWID + col] * 255.f, 0.f), 255.f));
                v = rintf(0.299f * fr + 0.587f * fg + 0.114f * fb) * (1.0f / 255.0f);
            } else {
                v = base[(size_t)r * WWID + col];
            }
            s_tile[lr][col] = v;
        }
        __syncthreads();
        // ---- 5x5 blur + noise stats for STRIP output rows ----
        for (int idx = tid; idx < STRIP * WWID; idx += 256) {
            const int ly = idx >> 9;
            const int xc = idx & 511;
            float blur = 0.f;
#pragma unroll
            for (int dy = 0; dy < 5; ++dy) {
                const float* row = s_tile[ly + dy];
                float rs = 0.f;
#pragma unroll
                for (int dx = 0; dx < 5; ++dx) {
                    int cx = xc + dx - 2;
                    cx = (cx < 0) ? -cx : (cx > 511 ? 1022 - cx : cx);
                    rs += gk[dx] * row[cx];
                }
                blur += gk[dy] * rs;
            }
            const float v = s_tile[ly + 2][xc] - blur;
            lsum += v;
            lsq  += v * v;
            if (GRAY) {
                if (v >= -0.5f && v <= 0.5f) {   // jnp.histogram range semantics
                    int bin = (int)((v + 0.5f) * 50.0f);
                    if (bin > 49) bin = 49;
                    atomicAdd(&s_hist[bin], 1u);
                }
            }
        }
    }

    // ---- wave32 shuffle reduction, then 8-partial LDS reduction ----
    float s = lsum, q = lsq;
#pragma unroll
    for (int off = 16; off > 0; off >>= 1) {
        s += __shfl_down(s, off, 32);
        q += __shfl_down(q, off, 32);
    }
    __syncthreads();
    if ((tid & 31) == 0) { s_psum[tid >> 5] = s; s_psq[tid >> 5] = q; }
    __syncthreads();
    if (tid == 0) {
        float ts = 0.f, tq = 0.f;
#pragma unroll
        for (int i = 0; i < 8; ++i) { ts += s_psum[i]; tq += s_psq[i]; }
        if (GRAY) { ws[WS_GSUM + blockIdx.x] = ts; ws[WS_GSQ + blockIdx.x] = tq; }
        else      { ws[WS_CHSUM + blockIdx.x] = ts; ws[WS_CHSQ + blockIdx.x] = tq; }
    }
    if (GRAY) {
        __syncthreads();
        for (int i = tid; i < 50; i += 256)
            ws[WS_HIST + blockIdx.x * 50 + i] = (float)s_hist[i];
    }
}

__global__ __launch_bounds__(512)
void mlp_kernel(const float* __restrict__ ws,
                const float* __restrict__ W1, const float* __restrict__ b1,
                const float* __restrict__ W2, const float* __restrict__ b2,
                float* __restrict__ out) {
    __shared__ float    s_feat[64][6];    // [B,6] features
    __shared__ _Float16 s_h[64][32];      // layer-1 activations (A matrix)
    __shared__ _Float16 s_w2t[32][64];    // W2^T  (B matrix, K x N)

    const int t = threadIdx.x;
    const float invN = 1.0f / (float)HW;

    // ---- stage A: finalize 6 features per image ----
    if (t < 64) {
        const int m = t;
        const float gs = ws[WS_GSUM + m], gq = ws[WS_GSQ + m];
        const float gmean = gs * invN;
        const float gvar  = fmaxf(gq * invN - gmean * gmean, 0.f);
        const float* h = ws + WS_HIST + m * 50;
        float total = 0.f;
        for (int i = 0; i < 50; ++i) total += h[i];
        const float itot = 1.0f / (total + 1e-10f);
        float ent = 0.f;
        for (int i = 0; i < 50; ++i) {
            const float p = h[i] * itot;
            ent -= p * log2f(p + 1e-10f);
        }
        s_feat[m][0] = gvar;          // local_var
        s_feat[m][1] = sqrtf(gvar);   // noise_std == sqrt(var)
        s_feat[m][2] = ent;           // histogram entropy
#pragma unroll
        for (int c = 0; c < 3; ++c) {
            const float cs = ws[WS_CHSUM + m * 3 + c];
            const float cq = ws[WS_CHSQ + m * 3 + c];
            const float cm = cs * invN;
            s_feat[m][3 + c] = sqrtf(fmaxf(cq * invN - cm * cm, 0.f));
        }
    }
    // stage W2^T into LDS as f16 (w2t[k][n] = W2[n,k])
    for (int i = t; i < 2048; i += 512) {
        const int k = i >> 6, n = i & 63;
        s_w2t[k][n] = (_Float16)W2[n * 32 + k];
    }
    __syncthreads();

    // ---- layer 1 (K=6, scalar): h = relu(feats @ W1^T + b1) ----
    for (int i = t; i < 2048; i += 512) {
        const int m = i >> 5, j = i & 31;
        float acc = b1[j];
#pragma unroll
        for (int k = 0; k < 6; ++k) acc += s_feat[m][k] * W1[j * 6 + k];
        s_h[m][j] = (_Float16)fmaxf(acc, 0.f);
    }
    __syncthreads();

    // ---- layer 2 via WMMA: out = relu(h @ W2^T + b2); M=N=64, K=32 ----
    // 16 waves; wave w handles tile (w/4, w%4). EXEC all ones here.
    const int lane = t & 31;
    const int wid  = t >> 5;
    const int tm   = (wid >> 2) * 16;
    const int tn   = (wid & 3) * 16;
    const int half = (lane < 16) ? 0 : 1;

    // A 16x32 f16, contiguous k order used here (matches B): lane holds row M=lane%16;
    //   lanes 0-15: K = 0..15 ; lanes 16-31: K = 16..31
    const int arow = tm + (lane & 15);
    const int ka0  = half ? 16 : 0;
    const int ka1  = half ? 24 : 8;
    v16h a;
#pragma unroll
    for (int i = 0; i < 8; ++i) {
        a[i]     = s_h[arow][ka0 + i];
        a[i + 8] = s_h[arow][ka1 + i];
    }
    // B 32x16 f16, contiguous k order used here (matches A): lane holds col N=lane%16;
    //   lanes 0-15: K = 0..15 ; lanes 16-31: K = 16..31
    const int bcol = tn + (lane & 15);
    const int kb   = half ? 16 : 0;
    v16h bb;
#pragma unroll
    for (int i = 0; i < 16; ++i) bb[i] = s_w2t[kb + i][bcol];

    v8f c = {};
    c = __builtin_amdgcn_wmma_f32_16x16x32_f16(
            /*neg_a=*/false, a, /*neg_b=*/false, bb,
            /*c_mod=*/(short)0, c, /*reuse_a=*/false, /*reuse_b=*/false);

    // C/D layout: VGPR r -> M = r (lanes 0-15) or r+8 (lanes 16-31), N = lane%16
    const int mbase = tm + (half ? 8 : 0);
    const int ncol  = tn + (lane & 15);
#pragma unroll
    for (int r = 0; r < 8; ++r) {
        out[(mbase + r) * 64 + ncol] = fmaxf(c[r] + b2[ncol], 0.f);
    }
}

extern "C" void kernel_launch(void* const* d_in, const int* in_sizes, int n_in,
                              void* d_out, int out_size, void* d_ws, size_t ws_size,
                              hipStream_t stream) {
    const float* x  = (const float*)d_in[0];  // [64,3,512,512]
    const float* W1 = (const float*)d_in[1];  // [32,6]
    const float* b1 = (const float*)d_in[2];  // [32]
    const float* W2 = (const float*)d_in[3];  // [64,32]
    const float* b2 = (const float*)d_in[4];  // [64]
    float* ws  = (float*)d_ws;
    float* out = (float*)d_out;               // [64,64]

    // Pass 1: 192 channel planes (per-channel noise std numerators)
    noise_stats_kernel<false><<<192, 256, 0, stream>>>(x, ws);
    // Pass 2: 64 images (gray noise sum/sumsq + 50-bin histogram);
    //         x likely still resident in the 192MB L2 from pass 1.
    noise_stats_kernel<true><<<64, 256, 0, stream>>>(x, ws);
    // Pass 3: features + MLP (layer 2 uses v_wmma_f32_16x16x32_f16)
    mlp_kernel<<<1, 512, 0, stream>>>(ws, W1, b1, W2, b2, out);
}
